// DecoderSeq_66907000537674
// MI455X (gfx1250) — compile-verified, matrix operand order checked by reading, not run on hardware
//
#include <hip/hip_runtime.h>
#include <hip/hip_bf16.h>

// ---------------------------------------------------------------------------
// LSTM decoder for MI455X (gfx1250).
// B=256, E=512, HID=1024, STEPS=256.
// Recurrent GEMM [256x1024]@[1024x4096] done with v_wmma_f32_16x16x32_bf16,
// W_hh slice resident in LDS for all 256 steps, c-state in registers,
// device-wide step barrier via atomics in workspace.
// ---------------------------------------------------------------------------

#define B_      256
#define E_      512
#define HID_    1024
#define GATES_  4096      // 4*HID
#define STEPS_  256
#define NWG_    64        // persistent workgroups; each owns 16 hidden units
#define TPB_    256       // 8 wave32 waves
#define LDS_WSTRIDE 1032  // 1024 + 8 element pad -> 2064 B rows, bank-conflict free

typedef __attribute__((ext_vector_type(16))) __bf16 v16bf;
typedef __attribute__((ext_vector_type(8)))  float  v8f;

union Frag {          // one WMMA 16x32 bf16 operand (8 VGPRs)
    v16bf f;
    uint4 q[2];
};

__device__ __forceinline__ float fast_rcp(float x) {
#if __has_builtin(__builtin_amdgcn_rcpf)
    return __builtin_amdgcn_rcpf(x);
#else
    return 1.0f / x;
#endif
}
__device__ __forceinline__ float fast_sigmoid(float x) {
    return fast_rcp(1.0f + __expf(-x));
}
__device__ __forceinline__ float fast_tanh(float x) {
#if __has_builtin(__builtin_amdgcn_tanhf)
    return __builtin_amdgcn_tanhf(x);
#else
    float e = __expf(2.0f * x);
    return 1.0f - 2.0f * fast_rcp(e + 1.0f);
#endif
}

// Device-wide generation barrier (workspace-backed, zeroed each launch).
__device__ __forceinline__ void grid_barrier(unsigned* cnt, unsigned* gen,
                                             unsigned target) {
    __syncthreads();
    if (threadIdx.x == 0) {
        __threadfence();
        unsigned prev = __hip_atomic_fetch_add(cnt, 1u, __ATOMIC_ACQ_REL,
                                               __HIP_MEMORY_SCOPE_AGENT);
        if (prev == (unsigned)(NWG_ - 1)) {
            __hip_atomic_store(cnt, 0u, __ATOMIC_RELAXED, __HIP_MEMORY_SCOPE_AGENT);
            __hip_atomic_fetch_add(gen, 1u, __ATOMIC_RELEASE, __HIP_MEMORY_SCOPE_AGENT);
        } else {
            while (__hip_atomic_load(gen, __ATOMIC_ACQUIRE,
                                     __HIP_MEMORY_SCOPE_AGENT) < target) {
                __builtin_amdgcn_s_sleep(4);
            }
        }
    }
    __syncthreads();
    __threadfence();
}

// --- prep: zero h double-buffer + barrier words --------------------------
__global__ void init_state_kernel(uint4* hbuf_vec, unsigned* bar) {
    int i = blockIdx.x * blockDim.x + threadIdx.x;   // 65536 uint4 = 1 MB
    hbuf_vec[i] = make_uint4(0u, 0u, 0u, 0u);
    if (i == 0) { bar[0] = 0u; bar[1] = 0u; }
}

// --- prep: W_hh fp32 -> bf16 (same [4096][1024] row-major layout) --------
__global__ void cvt_whh_kernel(const float* __restrict__ w,
                               __bf16* __restrict__ o) {
    int i = (blockIdx.x * blockDim.x + threadIdx.x) * 4;
    float4 v = *(const float4*)(w + i);
    ushort4 pk;
    pk.x = __builtin_bit_cast(unsigned short, (__bf16)v.x);
    pk.y = __builtin_bit_cast(unsigned short, (__bf16)v.y);
    pk.z = __builtin_bit_cast(unsigned short, (__bf16)v.z);
    pk.w = __builtin_bit_cast(unsigned short, (__bf16)v.w);
    *(ushort4*)((unsigned short*)o + i) = pk;
}

// --- prep: x_pre = z @ W_ih^T + b_ih + b_hh  (fp32, exact, once) ---------
__global__ void xpre_kernel(const float* __restrict__ z,
                            const float* __restrict__ wih,
                            const float* __restrict__ bih,
                            const float* __restrict__ bhh,
                            float* __restrict__ xpre) {
    int idx = blockIdx.x * blockDim.x + threadIdx.x;  // 1,048,576 threads
    int b = idx & (B_ - 1);
    int n = idx >> 8;                                  // 0..4095
    const float* zr = z + b * E_;
    const float* wr = wih + n * E_;
    float s = 0.0f;
    for (int k = 0; k < E_; k += 4) {
        float4 a = *(const float4*)(zr + k);
        float4 w = *(const float4*)(wr + k);
        s = fmaf(a.x, w.x, s);
        s = fmaf(a.y, w.y, s);
        s = fmaf(a.z, w.z, s);
        s = fmaf(a.w, w.w, s);
    }
    xpre[b * GATES_ + n] = s + bih[n] + bhh[n];
}

// --- main persistent LSTM kernel -----------------------------------------
__global__ __launch_bounds__(TPB_) void lstm_persistent(
        const float*  __restrict__ xpre,    // [256][4096]
        const __bf16* __restrict__ whh16,   // [4096][1024]
        const float*  __restrict__ wproj,   // [2][1024]
        const float*  __restrict__ bproj,   // [2]
        __bf16*       __restrict__ hbuf,    // 2 x [256][1024]
        float*        __restrict__ out,     // [256][256][2]
        unsigned*     __restrict__ bar) {
    extern __shared__ char smem[];
    __bf16* sW  = (__bf16*)smem;                       // [64][1032] padded rows
    float*  sWp = (float*)(smem + 64 * LDS_WSTRIDE * 2); // [2][1024]

    const int g    = blockIdx.x;        // 0..63, owns hidden units [16g,16g+16)
    const int tid  = threadIdx.x;
    const int wave = tid >> 5;
    const int lane = tid & 31;
    const int r    = lane & 15;
    const int half = lane >> 4;
    const int hidBase = g * 16;

    // Load this WG's 64 gate rows of W_hh (bf16) into LDS: rows stay resident
    // for all 256 steps -> zero weight bandwidth in the recurrent loop.
    for (int it = tid; it < 64 * 128; it += TPB_) {
        int row = it >> 7;            // 0..63 = gate*16 + jl
        int col = it & 127;           // uint4 column (8 elements each)
        int gate = row >> 4, jl = row & 15;
        const uint4* src = (const uint4*)(whh16 + (gate * HID_ + hidBase + jl) * HID_);
        *((uint4*)(sW + row * LDS_WSTRIDE) + col) = src[col];
    }
    for (int it = tid; it < 512; it += TPB_)
        ((uint4*)sWp)[it] = ((const uint4*)wproj)[it];
    __syncthreads();

    const int mBase = wave * 32;      // each wave owns 32 batch rows
    float c[2][8];                    // fp32 cell state, register-resident
#pragma unroll
    for (int i = 0; i < 2; ++i)
#pragma unroll
        for (int p = 0; p < 8; ++p) c[i][p] = 0.0f;

    for (int t = 0; t < STEPS_; ++t) {
        const __bf16* hprev = hbuf + (size_t)(t & 1) * (B_ * HID_);
        __bf16*       hnext = hbuf + (size_t)((t + 1) & 1) * (B_ * HID_);

        v8f acc[2][4];
#pragma unroll
        for (int i = 0; i < 2; ++i)
#pragma unroll
            for (int gg = 0; gg < 4; ++gg)
                acc[i][gg] = (v8f){0.f, 0.f, 0.f, 0.f, 0.f, 0.f, 0.f, 0.f};

        // GEMM: gate tile [256 x 64] += h[256x1024] * W^T slice, K in 32-chunks.
        for (int k0 = 0; k0 < HID_; k0 += 32) {
            Frag a[2], b[4];
#pragma unroll
            for (int i = 0; i < 2; ++i) {
                const __bf16* p = hprev + (size_t)(mBase + i * 16 + r) * HID_
                                        + k0 + half * 8;
                a[i].q[0] = *(const uint4*)p;
                a[i].q[1] = *(const uint4*)(p + 16);
            }
#pragma unroll
            for (int gg = 0; gg < 4; ++gg) {
                const __bf16* p = sW + (gg * 16 + r) * LDS_WSTRIDE + k0 + half * 8;
                b[gg].q[0] = *(const uint4*)p;
                b[gg].q[1] = *(const uint4*)(p + 16);
            }
#pragma unroll
            for (int i = 0; i < 2; ++i)
#pragma unroll
                for (int gg = 0; gg < 4; ++gg)
                    acc[i][gg] = __builtin_amdgcn_wmma_f32_16x16x32_bf16(
                        false, a[i].f, false, b[gg].f,
                        (short)0, acc[i][gg], false, false);
        }

        // Elementwise gate math. C/D layout: VGPR p, lanes0-15 -> M=p,
        // lanes16-31 -> M=p+8; N = r. Each lane slot owns one (batch, j).
        const int j = hidBase + r;
#pragma unroll
        for (int i = 0; i < 2; ++i) {
#pragma unroll
            for (int p = 0; p < 8; ++p) {
                int bb = mBase + i * 16 + half * 8 + p;
                const float* xp = xpre + (size_t)bb * GATES_ + j;
                float gi = fast_sigmoid(acc[i][0][p] + xp[0]);
                float gf = fast_sigmoid(acc[i][1][p] + xp[HID_]);
                float gc = fast_tanh   (acc[i][2][p] + xp[2 * HID_]);
                float go = fast_sigmoid(acc[i][3][p] + xp[3 * HID_]);
                float cn = gf * c[i][p] + gi * gc;
                c[i][p] = cn;
                float hn = go * fast_tanh(cn);
                hnext[(size_t)bb * HID_ + j] = (__bf16)hn;
            }
        }

        // All WGs must finish writing h(t) before anyone reads it.
        grid_barrier(bar, bar + 1, (unsigned)(t + 1));

        // Projection out[:, t, :]: WG g handles batches 4g..4g+3, wave->(b,p).
        {
            int bb   = g * 4 + (wave >> 1);
            int pidx = wave & 1;
            const __bf16* hr = hnext + (size_t)bb * HID_;
            const float*  wp = sWp + pidx * HID_;
            float s = 0.0f;
            for (int jj = lane; jj < HID_; jj += 32)
                s = fmaf((float)hr[jj], wp[jj], s);
#pragma unroll
            for (int off = 16; off > 0; off >>= 1)
                s += __shfl_xor(s, off, 32);
            if (lane == 0)
                out[(size_t)bb * (STEPS_ * 2) + t * 2 + pidx] =
                    fast_tanh(s + bproj[pidx]);
        }
    }
}

// ---------------------------------------------------------------------------
extern "C" void kernel_launch(void* const* d_in, const int* in_sizes, int n_in,
                              void* d_out, int out_size, void* d_ws, size_t ws_size,
                              hipStream_t stream) {
    const float* z     = (const float*)d_in[0];
    const float* wih   = (const float*)d_in[1];
    const float* whh   = (const float*)d_in[2];
    const float* bih   = (const float*)d_in[3];
    const float* bhh   = (const float*)d_in[4];
    const float* wproj = (const float*)d_in[5];
    const float* bproj = (const float*)d_in[6];
    float* out = (float*)d_out;

    // Workspace layout (needs ~13.7 MB):
    char* ws = (char*)d_ws;
    float*    xpre  = (float*)ws;                      //  4,194,304 B
    __bf16*   whh16 = (__bf16*)(ws + 4194304);         //  8,388,608 B
    __bf16*   hbuf  = (__bf16*)(ws + 12582912);        //  1,048,576 B (2 bufs)
    unsigned* bar   = (unsigned*)(ws + 13631488);      //  8 B

    // Re-init every call (graph replay determinism).
    init_state_kernel<<<256, 256, 0, stream>>>((uint4*)hbuf, bar);
    cvt_whh_kernel<<<GATES_ * HID_ / (256 * 4), 256, 0, stream>>>(whh, whh16);
    xpre_kernel<<<B_ * GATES_ / 256, 256, 0, stream>>>(z, wih, bih, bhh, xpre);

    const int ldsBytes = 64 * LDS_WSTRIDE * 2 + 2 * HID_ * 4;  // 140,288 B
    lstm_persistent<<<NWG_, TPB_, ldsBytes, stream>>>(
        xpre, whh16, wproj, bproj, hbuf, out, bar);
}
